// GraphAttentionLayer_59193239274119
// MI455X (gfx1250) — matrix operand order repaired and checked by reading, not compiled, not run on hardware
//
#include <hip/hip_runtime.h>
#include <hip/hip_bf16.h>

// ---------------------------------------------------------------------------
// GAT layer, fused flash-softmax formulation for MI455X (gfx1250, wave32).
//   N = 8192 nodes, F_IN = 128, F_OUT = 64.
//   adj stream (256 MB) read exactly once; P @ Wh done with
//   v_wmma_f32_16x16x32_bf16 under the memory stream.
// ---------------------------------------------------------------------------

#define GN    8192
#define GFIN  128
#define GFOUT 64
#define NEG_INF_F (-9.0e15f)

typedef __attribute__((ext_vector_type(16))) __bf16 v16bf;
typedef __attribute__((ext_vector_type(8)))  float  v8f;
typedef __attribute__((ext_vector_type(4)))  float  v4f;
typedef __attribute__((ext_vector_type(4)))  int    v4i;

// ---------------------------------------------------------------------------
// Kernel 1: Wh = x @ W^T   (8192x128 * 128x64 -> 8192x64), fp32.
// One thread per output element; x and W are L2-resident.
// ---------------------------------------------------------------------------
__global__ __launch_bounds__(256) void wh_kernel(const float* __restrict__ x,
                                                 const float* __restrict__ W,
                                                 float* __restrict__ Wh) {
    int idx = blockIdx.x * 256 + threadIdx.x;          // 0 .. 8192*64-1
    int r = idx >> 6;
    int c = idx & 63;
    const float* xr = x + (size_t)r * GFIN;
    const float* wr = W + (size_t)c * GFIN;
    float s = 0.f;
#pragma unroll
    for (int k = 0; k < GFIN; k += 4) {
        v4f a = *(const v4f*)(xr + k);
        v4f b = *(const v4f*)(wr + k);
        s += a.x * b.x + a.y * b.y + a.z * b.z + a.w * b.w;
    }
    Wh[idx] = s;
}

// ---------------------------------------------------------------------------
// Kernel 2: Wh1 = Wh @ a1, Wh2 = Wh @ a2 (fp32), and WhbT[c][i] = bf16(Wh[i][c])
// (transposed bf16 copy so WMMA B operands are K-contiguous 32B loads).
// ---------------------------------------------------------------------------
__global__ __launch_bounds__(256) void prep_kernel(const float* __restrict__ Wh,
                                                   const float* __restrict__ a,
                                                   float* __restrict__ Wh1,
                                                   float* __restrict__ Wh2,
                                                   __bf16* __restrict__ WhbT) {
    int i = blockIdx.x * 256 + threadIdx.x;            // 0 .. 8191
    const float* whr = Wh + (size_t)i * GFOUT;
    float s1 = 0.f, s2 = 0.f;
#pragma unroll
    for (int c = 0; c < GFOUT; ++c) {
        float v = whr[c];
        s1 += v * a[c];
        s2 += v * a[GFOUT + c];
        WhbT[(size_t)c * GN + i] = (__bf16)v;          // coalesced across threads
    }
    Wh1[i] = s1;
    Wh2[i] = s2;
}

// ---------------------------------------------------------------------------
// Kernel 3: fused masked-softmax attention + P@Wh, online softmax.
// Block = 16 output rows, 4 waves; wave w streams j-tiles w,w+4,w+8,... of 64
// columns each (adj read exactly once per element). Cross-wave merge in LDS.
// ---------------------------------------------------------------------------
__global__ __launch_bounds__(128) void gat_attn_kernel(const int*    __restrict__ adj,
                                                       const __bf16* __restrict__ WhbT,
                                                       const float*  __restrict__ Wh1,
                                                       const float*  __restrict__ Wh2,
                                                       float*        __restrict__ out) {
    const int wave  = threadIdx.x >> 5;   // 0..3
    const int lane  = threadIdx.x & 31;
    const int row16 = lane & 15;          // A-matrix row (and B/C column) id
    const int half  = lane >> 4;          // 0 or 1
    const int kbase = half * 8;           // 16-bit A-matrix K base per ISA layout
    const int i0    = blockIdx.x * 16;

    __shared__ float lds_acc[4][16][64];
    __shared__ float lds_m[4][16];
    __shared__ float lds_l[4][16];

    const float wh1 = Wh1[i0 + row16];
    const int* adjRow = adj + (size_t)(i0 + row16) * GN;

    v8f acc[4];
#pragma unroll
    for (int t = 0; t < 4; ++t)
#pragma unroll
        for (int v = 0; v < 8; ++v) acc[t][v] = 0.f;

    float m = -3.0e38f;   // running row max
    float l = 0.f;        // running row sum

    for (int j0 = wave * 64; j0 < GN; j0 += 256) {
        // Prefetch next adj tile for this wave (global_prefetch_b8).
        __builtin_prefetch((const void*)(adjRow + j0 + 256), 0, 1);

        // ---- compute masked leaky-relu logits for this lane's 32 K values ---
        // A-matrix 16x32 bf16 layout: lane holds K contiguous
        //   [16*half..16*half+15] per 32-wide K half (matches B).
        float e[32];
        float tmax = NEG_INF_F;
#pragma unroll
        for (int h = 0; h < 2; ++h) {
#pragma unroll
            for (int rr = 0; rr < 2; ++rr) {
                const int k0 = 32 * h + 16 * half + 8 * rr;
                v4i a0 = *(const v4i*)(adjRow + j0 + k0);
                v4i a1 = *(const v4i*)(adjRow + j0 + k0 + 4);
                v4f w0 = *(const v4f*)(Wh2 + j0 + k0);
                v4f w1 = *(const v4f*)(Wh2 + j0 + k0 + 4);
#pragma unroll
                for (int q = 0; q < 8; ++q) {
                    float z  = wh1 + (q < 4 ? w0[q] : w1[q - 4]);
                    float ev = (z > 0.f) ? z : 0.2f * z;              // leaky_relu(0.2)
                    int   av = (q < 4 ? a0[q] : a1[q - 4]);
                    ev = (av > 0) ? ev : NEG_INF_F;                   // adj mask
                    e[16 * h + 8 * rr + q] = ev;
                    tmax = fmaxf(tmax, ev);
                }
            }
        }

        // ---- online softmax bookkeeping (row split across lane, lane^16) ----
        tmax = fmaxf(tmax, __shfl_xor(tmax, 16, 32));
        const float m_new = fmaxf(m, tmax);
        const float scale = __expf(m - m_new);

        float lsum = 0.f;
#pragma unroll
        for (int q = 0; q < 32; ++q) {
            float pv = __expf(e[q] - m_new);   // masked entries -> 0
            e[q] = pv;
            lsum += pv;
        }
        lsum += __shfl_xor(lsum, 16, 32);
        l = l * scale + lsum;
        m = m_new;

        // Rescale accumulator rows: C-layout row = v + 8*half; lane r holds
        // scale for row r (r = lane&15), so ds_bpermute broadcasts it.
#pragma unroll
        for (int v = 0; v < 8; ++v) {
            float s = __shfl(scale, v + 8 * half, 32);
#pragma unroll
            for (int t = 0; t < 4; ++t) acc[t][v] *= s;
        }

        // ---- pack A operands (two 16x32 bf16 tiles) ------------------------
        v16bf aop0, aop1;
#pragma unroll
        for (int q = 0; q < 16; ++q) {
            aop0[q] = (__bf16)e[q];
            aop1[q] = (__bf16)e[16 + q];
        }

        // ---- B operands + WMMA accumulate ----------------------------------
        // B 32x16 bf16 layout: lane column = lane&15, K = 16*half + 0..15
        // (contiguous) -> single 32-byte load from the transposed bf16 Wh.
        const __bf16* bbase = WhbT + (size_t)row16 * GN + j0 + 16 * half;
#pragma unroll
        for (int t = 0; t < 4; ++t) {
            const __bf16* bp = bbase + (size_t)(16 * t) * GN;
            v16bf b0 = *(const v16bf*)(bp);         // K half 0
            v16bf b1 = *(const v16bf*)(bp + 32);    // K half 1
            acc[t] = __builtin_amdgcn_wmma_f32_16x16x32_bf16(
                false, aop0, false, b0, (short)0, acc[t], false, false);
            acc[t] = __builtin_amdgcn_wmma_f32_16x16x32_bf16(
                false, aop1, false, b1, (short)0, acc[t], false, false);
        }
    }

    // ---- spill per-wave partials and merge the 4 softmax streams -----------
#pragma unroll
    for (int t = 0; t < 4; ++t)
#pragma unroll
        for (int v = 0; v < 8; ++v)
            lds_acc[wave][v + 8 * half][16 * t + row16] = acc[t][v];
    if (half == 0) {
        lds_m[wave][row16] = m;
        lds_l[wave][row16] = l;
    }
    __syncthreads();

    for (int idx = threadIdx.x; idx < 16 * 64; idx += 128) {
        const int r = idx >> 6;
        const int c = idx & 63;
        float mM = lds_m[0][r];
#pragma unroll
        for (int w = 1; w < 4; ++w) mM = fmaxf(mM, lds_m[w][r]);
        float L = 0.f, A = 0.f;
#pragma unroll
        for (int w = 0; w < 4; ++w) {
            float s = __expf(lds_m[w][r] - mM);
            L += lds_l[w][r] * s;
            A += lds_acc[w][r][c] * s;
        }
        float h = A / L;                               // softmax-normalized row
        out[(size_t)(i0 + r) * GFOUT + c] = (h > 0.f) ? h : (__expf(h) - 1.f);  // elu
    }
}

// ---------------------------------------------------------------------------
extern "C" void kernel_launch(void* const* d_in, const int* in_sizes, int n_in,
                              void* d_out, int out_size, void* d_ws, size_t ws_size,
                              hipStream_t stream) {
    const float* x   = (const float*)d_in[0];   // (8192,128) f32
    const int*   adj = (const int*)  d_in[1];   // (8192,8192) i32
    const float* W   = (const float*)d_in[2];   // (64,128) f32
    const float* a   = (const float*)d_in[3];   // (1,128) f32
    float*       out = (float*)d_out;           // (8192,64) f32

    char* ws = (char*)d_ws;
    float*  Wh   = (float*)(ws);                                  // 2 MB
    __bf16* WhbT = (__bf16*)(ws + (size_t)2 * 1024 * 1024);       // 1 MB
    float*  Wh1  = (float*)(ws + (size_t)3 * 1024 * 1024);        // 32 KB
    float*  Wh2  = (float*)(ws + (size_t)3 * 1024 * 1024 + 32768);// 32 KB

    wh_kernel  <<<(GN * GFOUT) / 256, 256, 0, stream>>>(x, W, Wh);
    prep_kernel<<<GN / 256,           256, 0, stream>>>(Wh, a, Wh1, Wh2, WhbT);
    gat_attn_kernel<<<GN / 16,        128, 0, stream>>>(adj, WhbT, Wh1, Wh2, out);
}
